// Mab_27058293965219
// MI455X (gfx1250) — matrix operand order repaired and checked by reading, not compiled, not run on hardware
//
#include <hip/hip_runtime.h>
#include <hip/hip_bf16.h>
#include <math.h>
#include <stdint.h>

// ---------------------------------------------------------------------------
// MI455X (gfx1250) MAB-gMLP block.
//  - GEMMs: v_wmma_f32_16x16x32_bf16 (bf16 operands, fp32 accumulate)
//  - Staging: global_load_async_to_lds_b128 (ASYNCcnt) with double-buffered
//    weight tiles (software pipeline), LDS fragment reads via ds_load_b128.
//  - LN / GELU / gating in fp32.
// ---------------------------------------------------------------------------

typedef __attribute__((ext_vector_type(16))) __bf16 v16bf;
typedef __attribute__((ext_vector_type(8)))  float  v8f;

#define BB   2
#define HH   256
#define WW   256
#define CC   128
#define GG   256          // heads per image
#define NN   256          // patches per head
#define CI   64
#define F1   384
#define HF   192
#define BG   512          // B * G
#define EPS  1e-3f

__device__ __forceinline__ unsigned short f32_to_bf16(float f) {
  unsigned int u = __float_as_uint(f);
  unsigned int r = u + 0x7FFFu + ((u >> 16) & 1u);   // round-to-nearest-even
  return (unsigned short)(r >> 16);
}
__device__ __forceinline__ float bf16_to_f32(unsigned short h) {
  return __uint_as_float(((unsigned int)h) << 16);
}
__device__ __forceinline__ float gelu_exact(float x) {
  return 0.5f * x * (1.0f + erff(x * 0.70710678118654752f));
}
__device__ __forceinline__ v8f wmma_bf16(v16bf a, v16bf b, v8f c) {
  return __builtin_amdgcn_wmma_f32_16x16x32_bf16(false, a, false, b,
                                                 (short)0, c, false, false);
}

// ---- CDNA5 async global->LDS copy (ASYNCcnt-tracked), 16B per lane --------
__device__ __forceinline__ void async_b128(const void* gsrc, void* lds_dst) {
  // generic pointer into LDS: low 32 bits are the LDS byte offset
  asm volatile("global_load_async_to_lds_b128 %0, %1, off"
               :: "v"((unsigned)(uintptr_t)lds_dst),
                  "v"((unsigned long long)(uintptr_t)gsrc)
               : "memory");
}
template <int N>
__device__ __forceinline__ void wait_asynccnt() {
  asm volatile("s_wait_asynccnt %0" :: "i"(N) : "memory");
}

// ---- WMMA fragment loads from LDS (ISA 7.12.2 layouts, wave32) -------------
template <int LD>
__device__ __forceinline__ v16bf load_frag_a(const unsigned short* lds,
                                             int m0, int kk) {
  int lane = threadIdx.x & 31;
  int m = lane & 15, half = lane >> 4;
  const unsigned short* rowp = lds + (size_t)(m0 + m) * LD + kk;
  union { uint4 q[2]; v16bf v; } f;
  f.q[0] = *(const uint4*)(rowp + half * 8);
  f.q[1] = *(const uint4*)(rowp + 16 + half * 8);
  return f.v;
}
template <int LD>
__device__ __forceinline__ v16bf load_frag_b(const unsigned short* ldsT,
                                             int n0, int kk) {
  int lane = threadIdx.x & 31;
  int nn = lane & 15, half = lane >> 4;
  const unsigned short* rowp = ldsT + (size_t)(n0 + nn) * LD + kk;
  union { uint4 q[2]; v16bf v; } f;
  f.q[0] = *(const uint4*)(rowp + half * 8);
  f.q[1] = *(const uint4*)(rowp + 16 + half * 8);
  return f.v;
}

// ===========================================================================
// K0: one-shot weight convert+transpose to bf16 (GEMM-ready layouts).
// ===========================================================================
__global__ __launch_bounds__(256) void k0_wpt(const float* __restrict__ Wp,
                                              unsigned short* __restrict__ o) {
  int i = blockIdx.x * 256 + threadIdx.x;          // i = n*128 + k
  int n = i >> 7, k = i & 127;
  o[i] = f32_to_bf16(Wp[(size_t)k * CC + n]);
}
__global__ __launch_bounds__(256) void k0_w1t(const float* __restrict__ W1,
                                              unsigned short* __restrict__ o) {
  size_t i = (size_t)blockIdx.x * 256 + threadIdx.x;  // ((n*384+f)*64)+k
  int k = (int)(i & 63);
  size_t t = i >> 6;
  int f = (int)(t % F1), n = (int)(t / F1);
  o[i] = f32_to_bf16(W1[((size_t)n * CI + k) * F1 + f]);
}
__global__ __launch_bounds__(256) void k0_wsgut(const float* __restrict__ Ws,
                                                unsigned short* __restrict__ o) {
  size_t i = (size_t)blockIdx.x * 256 + threadIdx.x;  // ((c*256+m)*256)+nn
  int nn = (int)(i & 255);
  size_t t = i >> 8;
  int m = (int)(t & 255), c = (int)(t >> 8);
  o[i] = f32_to_bf16(Ws[((size_t)c * NN + nn) * NN + m]);
}
__global__ __launch_bounds__(256) void k0_w2t(const float* __restrict__ W2,
                                              unsigned short* __restrict__ o) {
  size_t i = (size_t)blockIdx.x * 256 + threadIdx.x;  // ((n*64+f)*192)+k
  int k = (int)(i % HF);
  size_t t = i / HF;
  int f = (int)(t & 63), n = (int)(t >> 6);
  o[i] = f32_to_bf16(W2[((size_t)n * HF + k) * CI + f]);
}

// ===========================================================================
// K1: LN1 + proj GEMM [131072x128]x[128x128] + GELU + blockify split.
// W_proj^T (bf16) async-loaded whole; A tile LN'd in fp32, K chunked 64.
// ===========================================================================
#define LDA1 72
#define LDB1 136
__global__ __launch_bounds__(256) void k1_ln1_proj(
    const float* __restrict__ x, const float* __restrict__ ln1_g,
    const float* __restrict__ ln1_b, const unsigned short* __restrict__ Wpt,
    const float* __restrict__ bp, float* __restrict__ xb) {
  __shared__ __align__(16) unsigned short Al[128 * LDA1];
  __shared__ __align__(16) unsigned short Bt[128 * LDB1];
  const int tid = threadIdx.x;
  const int row0 = blockIdx.x * 128;

  // async-load the full 128x128 bf16 W_proj^T tile (8 x 16B chunks / thread)
#pragma unroll
  for (int j = 0; j < 8; ++j) {
    int cch = tid * 8 + j;
    int r = cch >> 4, piece = cch & 15;
    async_b128(Wpt + (size_t)r * CC + piece * 8,
               Bt + (size_t)r * LDB1 + piece * 8);
  }

  float mean = 0.f, rstd = 0.f;
  if (tid < 128) {
    const float* xr = x + (size_t)(row0 + tid) * CC;
    float s = 0.f, sq = 0.f;
#pragma unroll 4
    for (int k = 0; k < CC; ++k) { float v = xr[k]; s += v; sq += v * v; }
    mean = s * (1.f / CC);
    rstd = rsqrtf(sq * (1.f / CC) - mean * mean + EPS);
  }

  v8f acc[8];
#pragma unroll
  for (int t = 0; t < 8; ++t)
#pragma unroll
    for (int i = 0; i < 8; ++i) acc[t][i] = 0.f;
  const int m0 = (tid >> 5) * 16;

  for (int kt = 0; kt < CC; kt += 64) {
    __syncthreads();                       // Al safe to overwrite
    if (tid < 128) {                       // LN'd A half-tile (bf16)
      const float* xr = x + (size_t)(row0 + tid) * CC + kt;
      unsigned short* dst = Al + (size_t)tid * LDA1;
#pragma unroll 4
      for (int k = 0; k < 64; ++k) {
        float v = (xr[k] - mean) * rstd * ln1_g[kt + k] + ln1_b[kt + k];
        dst[k] = f32_to_bf16(v);
      }
    }
    wait_asynccnt<0>();                    // W tile resident (1st iter)
    __syncthreads();
#pragma unroll
    for (int kk = 0; kk < 64; kk += 32) {
      v16bf a = load_frag_a<LDA1>(Al, m0, kk);
#pragma unroll
      for (int t = 0; t < 8; ++t) {
        v16bf b = load_frag_b<LDB1>(Bt, t * 16, kt + kk);
        acc[t] = wmma_bf16(a, b, acc[t]);
      }
    }
  }

  const int lane = tid & 31;
#pragma unroll
  for (int t = 0; t < 8; ++t) {
#pragma unroll
    for (int v = 0; v < 8; ++v) {
      int rrow = m0 + ((lane < 16) ? v : v + 8);
      int col = t * 16 + (lane & 15);
      int R = row0 + rrow;
      float val = gelu_exact(acc[t][v] + bp[col]);
      int br = col >> 6, c = col & 63;
      int b = R >> 16, hw = R & 65535;
      int h = hw >> 8, w = hw & 255;
      int g = ((h >> 4) << 4) + (w >> 4);
      int n = ((h & 15) << 4) + (w & 15);
      int bg = b * GG + g;
      xb[(((size_t)br * BG + bg) * NN + n) * CI + c] = val;
    }
  }
}

// ===========================================================================
// K2: LN2 over CI=64, wave-per-row, fp32 -> bf16 GEMM operand.
// ===========================================================================
__global__ __launch_bounds__(256) void k2_ln2(
    const float* __restrict__ xb, const float* __restrict__ g,
    const float* __restrict__ b, unsigned short* __restrict__ a1) {
  const int wave = threadIdx.x >> 5, lane = threadIdx.x & 31;
  const size_t row = (size_t)blockIdx.x * 8 + wave;
  const float* p = xb + row * CI;
  float v0 = p[lane * 2], v1 = p[lane * 2 + 1];
  float s = v0 + v1, sq = v0 * v0 + v1 * v1;
#pragma unroll
  for (int m = 16; m >= 1; m >>= 1) {
    s += __shfl_xor(s, m, 32);
    sq += __shfl_xor(sq, m, 32);
  }
  float mean = s * (1.f / CI);
  float rstd = rsqrtf(sq * (1.f / CI) - mean * mean + EPS);
  unsigned short* o = a1 + row * CI;
  int c0 = lane * 2;
  o[c0]     = f32_to_bf16((v0 - mean) * rstd * g[c0] + b[c0]);
  o[c0 + 1] = f32_to_bf16((v1 - mean) * rstd * g[c0 + 1] + b[c0 + 1]);
}

// ===========================================================================
// K3: stage-1 batched GEMM: per patch n, [1024x64]x[64x384] + b1 + GELU.
// grid = (Mtile=8, Ftile=3, n=256); A and B^T tiles fully async-staged.
// ===========================================================================
#define LDK3 72
__global__ __launch_bounds__(256) void k3_stage1(
    const unsigned short* __restrict__ a1,
    const unsigned short* __restrict__ W1t, const float* __restrict__ b1,
    unsigned short* __restrict__ y) {
  __shared__ __align__(16) unsigned short Al[128 * LDK3];
  __shared__ __align__(16) unsigned short Bt[128 * LDK3];
  const int tid = threadIdx.x;
  const int row0 = blockIdx.x * 128;
  const int f0 = blockIdx.y * 128;
  const int n = blockIdx.z;
  {   // A tile: 128 rows x 64 bf16; 4 x 16B chunks per thread
    int r = tid >> 1, part = tid & 1;
    const unsigned short* src =
        a1 + ((size_t)(row0 + r) * NN + n) * CI + part * 32;
    unsigned short* dst = Al + (size_t)r * LDK3 + part * 32;
#pragma unroll
    for (int j = 0; j < 4; ++j) async_b128(src + j * 8, dst + j * 8);
  }
  {   // B^T tile: rows f_local, contiguous k (bf16 W1t)
    int fl = tid & 127, half = tid >> 7;
    const unsigned short* src =
        W1t + ((size_t)n * F1 + f0 + fl) * CI + half * 32;
    unsigned short* dst = Bt + (size_t)fl * LDK3 + half * 32;
#pragma unroll
    for (int j = 0; j < 4; ++j) async_b128(src + j * 8, dst + j * 8);
  }
  wait_asynccnt<0>();
  __syncthreads();

  v8f acc[8];
#pragma unroll
  for (int t = 0; t < 8; ++t)
#pragma unroll
    for (int i = 0; i < 8; ++i) acc[t][i] = 0.f;
  const int m0 = (tid >> 5) * 16;
#pragma unroll
  for (int kk = 0; kk < 64; kk += 32) {
    v16bf a = load_frag_a<LDK3>(Al, m0, kk);
#pragma unroll
    for (int t = 0; t < 8; ++t) {
      v16bf b = load_frag_b<LDK3>(Bt, t * 16, kk);
      acc[t] = wmma_bf16(a, b, acc[t]);
    }
  }
  const int lane = tid & 31;
#pragma unroll
  for (int t = 0; t < 8; ++t) {
#pragma unroll
    for (int v = 0; v < 8; ++v) {
      int row = row0 + m0 + ((lane < 16) ? v : v + 8);
      int f = f0 + t * 16 + (lane & 15);
      float val = gelu_exact(acc[t][v] + b1[(size_t)n * F1 + f]);
      y[((size_t)row * NN + n) * F1 + f] = f32_to_bf16(val);
    }
  }
}

// ===========================================================================
// K4: LN3 over HF=192 on u half of y, write transposed u_t[row][c][n] bf16.
// ===========================================================================
__global__ __launch_bounds__(256) void k4_ln3_t(
    const unsigned short* __restrict__ y, const float* __restrict__ g,
    const float* __restrict__ b, unsigned short* __restrict__ u_t) {
  const int wave = threadIdx.x >> 5, lane = threadIdx.x & 31;
  const size_t row = (size_t)blockIdx.x * 8 + wave;   // row = r*NN + n
  const unsigned short* p = y + row * F1;
  float v[6];
  float s = 0.f, sq = 0.f;
#pragma unroll
  for (int i = 0; i < 6; ++i) {
    v[i] = bf16_to_f32(p[lane * 6 + i]);
    s += v[i]; sq += v[i] * v[i];
  }
#pragma unroll
  for (int m = 16; m >= 1; m >>= 1) {
    s += __shfl_xor(s, m, 32);
    sq += __shfl_xor(sq, m, 32);
  }
  float mean = s * (1.f / HF);
  float rstd = rsqrtf(sq * (1.f / HF) - mean * mean + EPS);
  int r = (int)(row >> 8), n = (int)(row & 255);
#pragma unroll
  for (int i = 0; i < 6; ++i) {
    int c = lane * 6 + i;
    float o = (v[i] - mean) * rstd * g[c] + b[c];
    u_t[((size_t)r * HF + c) * NN + n] = f32_to_bf16(o);
  }
}

// ===========================================================================
// K5: spatial gating: per channel c, [1024x256]x[256x256] + bsgu, gate by v.
// grid = (Mtile=8, OutTile=2, c=192); K chunked 64; DOUBLE-BUFFERED weight
// tiles: next B^T tile's async copies in flight during current compute
// (s_wait_asynccnt 4 completes current tiles, leaves prefetch pending).
// ===========================================================================
#define LDK5 72
__global__ __launch_bounds__(256) void k5_sgu(
    const unsigned short* __restrict__ u_t,
    const unsigned short* __restrict__ Wst, const float* __restrict__ bsgu,
    const unsigned short* __restrict__ y, unsigned short* __restrict__ pbuf) {
  __shared__ __align__(16) unsigned short Al[128 * LDK5];
  __shared__ __align__(16) unsigned short Bt[2][128 * LDK5];
  const int tid = threadIdx.x;
  const int row0 = blockIdx.x * 128;
  const int mcol0 = blockIdx.y * 128;
  const int c = blockIdx.z;

  const int fl = tid & 127, halfB = tid >> 7;
  const int rA = tid >> 1, pA = tid & 1;
  auto issueB = [&](int kt, int buf) {
    const unsigned short* src =
        Wst + ((size_t)c * NN + mcol0 + fl) * NN + kt + halfB * 32;
    unsigned short* dst = Bt[buf] + (size_t)fl * LDK5 + halfB * 32;
#pragma unroll
    for (int j = 0; j < 4; ++j) async_b128(src + j * 8, dst + j * 8);
  };
  issueB(0, 0);                            // prologue prefetch

  v8f acc[8];
#pragma unroll
  for (int t = 0; t < 8; ++t)
#pragma unroll
    for (int i = 0; i < 8; ++i) acc[t][i] = 0.f;
  const int m0 = (tid >> 5) * 16;

  for (int it = 0; it < 4; ++it) {
    const int kt = it * 64, cur = it & 1;
    __syncthreads();                       // prior reads of Al / Bt[next] done
    {   // A tile for this K chunk
      const unsigned short* src =
          u_t + ((size_t)(row0 + rA) * HF + c) * NN + kt + pA * 32;
      unsigned short* dst = Al + (size_t)rA * LDK5 + pA * 32;
#pragma unroll
      for (int j = 0; j < 4; ++j) async_b128(src + j * 8, dst + j * 8);
    }
    if (it + 1 < 4) issueB(kt + 64, cur ^ 1);
    wait_asynccnt<4>();                    // Bt[cur] + Al done (in-order cnt)
    __syncthreads();
#pragma unroll
    for (int kk = 0; kk < 64; kk += 32) {
      v16bf a = load_frag_a<LDK5>(Al, m0, kk);
#pragma unroll
      for (int t = 0; t < 8; ++t) {
        v16bf b = load_frag_b<LDK5>(Bt[cur], t * 16, kk);
        acc[t] = wmma_bf16(a, b, acc[t]);
      }
    }
  }
  const int lane = tid & 31;
#pragma unroll
  for (int t = 0; t < 8; ++t) {
#pragma unroll
    for (int v = 0; v < 8; ++v) {
      int row = row0 + m0 + ((lane < 16) ? v : v + 8);
      int m = mcol0 + t * 16 + (lane & 15);
      float gated = acc[t][v] + bsgu[(size_t)c * NN + m];
      float vv = bf16_to_f32(y[((size_t)row * NN + m) * F1 + HF + c]);
      pbuf[((size_t)row * NN + m) * HF + c] = f32_to_bf16(gated * vv);
    }
  }
}

// ===========================================================================
// K6: stage-2: per patch n, [1024x192]x[192x64] + b2 + gmlp residual +
// block/grid reconstruction + final x residual -> out, glob.
// grid = (Mtile=8, n=256); pipelined like K5 (Bt double-buffered).
// ===========================================================================
#define LDK6 72
__global__ __launch_bounds__(256) void k6_stage2(
    const unsigned short* __restrict__ pbuf,
    const unsigned short* __restrict__ W2t, const float* __restrict__ b2,
    const float* __restrict__ xb, const float* __restrict__ x,
    float* __restrict__ out, float* __restrict__ glob) {
  __shared__ __align__(16) unsigned short Al[128 * LDK6];
  __shared__ __align__(16) unsigned short Bt[2][64 * LDK6];
  const int tid = threadIdx.x;
  const int row0 = blockIdx.x * 128;
  const int n = blockIdx.y;

  const int flB = tid & 63, qB = tid >> 6;           // 64 rows x 4 quarters
  const int rA = tid >> 1, pA = tid & 1;
  auto issueB = [&](int kt, int buf) {
    const unsigned short* src =
        W2t + ((size_t)n * CI + flB) * HF + kt + qB * 16;
    unsigned short* dst = Bt[buf] + (size_t)flB * LDK6 + qB * 16;
#pragma unroll
    for (int j = 0; j < 2; ++j) async_b128(src + j * 8, dst + j * 8);
  };
  issueB(0, 0);

  v8f acc[4];
#pragma unroll
  for (int t = 0; t < 4; ++t)
#pragma unroll
    for (int i = 0; i < 8; ++i) acc[t][i] = 0.f;
  const int m0 = (tid >> 5) * 16;

  for (int it = 0; it < 3; ++it) {
    const int kt = it * 64, cur = it & 1;
    __syncthreads();
    {
      const unsigned short* src =
          pbuf + ((size_t)(row0 + rA) * NN + n) * HF + kt + pA * 32;
      unsigned short* dst = Al + (size_t)rA * LDK6 + pA * 32;
#pragma unroll
      for (int j = 0; j < 4; ++j) async_b128(src + j * 8, dst + j * 8);
    }
    if (it + 1 < 3) issueB(kt + 64, cur ^ 1);
    wait_asynccnt<2>();                    // Bt[cur]+Al done, prefetch pending
    __syncthreads();
#pragma unroll
    for (int kk = 0; kk < 64; kk += 32) {
      v16bf a = load_frag_a<LDK6>(Al, m0, kk);
#pragma unroll
      for (int t = 0; t < 4; ++t) {
        v16bf b = load_frag_b<LDK6>(Bt[cur], t * 16, kk);
        acc[t] = wmma_bf16(a, b, acc[t]);
      }
    }
  }
  const int lane = tid & 31;
#pragma unroll
  for (int t = 0; t < 4; ++t) {
#pragma unroll
    for (int v = 0; v < 8; ++v) {
      int row = row0 + m0 + ((lane < 16) ? v : v + 8);
      int f = t * 16 + (lane & 15);
      float val = acc[t][v] + b2[(size_t)n * CI + f] +
                  xb[((size_t)row * NN + n) * CI + f];
      int br = row >> 9, bg = row & 511;
      int b = bg >> 8, g = bg & 255;
      int gh = g >> 4, gw = g & 15;
      int bh = n >> 4, bw = n & 15;
      if (br == 0) {                 // local branch: recon_grids
        int h = bh * 16 + gh, w = bw * 16 + gw;
        size_t pix = ((size_t)b * HH + h) * WW + w;
        out[pix * CC + f] = val + x[pix * CC + f];
      } else {                       // glob branch: recon_blocks
        int h = gh * 16 + bh, w = gw * 16 + bw;
        size_t pix = ((size_t)b * HH + h) * WW + w;
        out[pix * CC + CI + f] = val + x[pix * CC + CI + f];
        glob[pix * CI + f] = val;    // second tuple output (no +x)
      }
    }
  }
}

// ===========================================================================
extern "C" void kernel_launch(void* const* d_in, const int* in_sizes, int n_in,
                              void* d_out, int out_size, void* d_ws,
                              size_t ws_size, hipStream_t stream) {
  const float* x      = (const float*)d_in[0];
  const float* ln1_g  = (const float*)d_in[1];
  const float* ln1_b  = (const float*)d_in[2];
  const float* W_proj = (const float*)d_in[3];
  const float* b_proj = (const float*)d_in[4];
  const float* ln2_g  = (const float*)d_in[5];
  const float* ln2_b  = (const float*)d_in[6];
  const float* W1     = (const float*)d_in[7];
  const float* b1     = (const float*)d_in[8];
  const float* ln3_g  = (const float*)d_in[9];
  const float* ln3_b  = (const float*)d_in[10];
  const float* Wsgu   = (const float*)d_in[11];
  const float* bsgu   = (const float*)d_in[12];
  const float* W2     = (const float*)d_in[13];
  const float* b2     = (const float*)d_in[14];

  float* out  = (float*)d_out;
  float* glob = out + (size_t)BB * HH * WW * CC;

  char* ws = (char*)d_ws;
  float*          xb    = (float*)ws;                              //  32 MB
  unsigned short* a1    = (unsigned short*)(ws + 33554432ull);     //  16 MB
  unsigned short* ybuf  = (unsigned short*)(ws + 50331648ull);     // 192 MB
  unsigned short* u_t   = (unsigned short*)(ws + 251658240ull);    //  96 MB
  unsigned short* pbuf  = (unsigned short*)(ws + 352321536ull);    //  96 MB
  unsigned short* Wpt   = (unsigned short*)(ws + 452984832ull);    //  32 KB
  unsigned short* W1t   = (unsigned short*)(ws + 453017600ull);    //  12 MB
  unsigned short* Wsgut = (unsigned short*)(ws + 465600512ull);    //  24 MB
  unsigned short* W2t   = (unsigned short*)(ws + 490766336ull);    //   6 MB

  // weight prep (bf16, transposed for contiguous async staging)
  k0_wpt  <<<64,    256, 0, stream>>>(W_proj, Wpt);
  k0_w1t  <<<24576, 256, 0, stream>>>(W1, W1t);
  k0_wsgut<<<49152, 256, 0, stream>>>(Wsgu, Wsgut);
  k0_w2t  <<<12288, 256, 0, stream>>>(W2, W2t);

  k1_ln1_proj<<<1024, 256, 0, stream>>>(x, ln1_g, ln1_b, Wpt, b_proj, xb);
  k2_ln2<<<32768, 256, 0, stream>>>(xb, ln2_g, ln2_b, a1);
  k3_stage1<<<dim3(8, 3, 256), 256, 0, stream>>>(a1, W1t, b1, ybuf);
  k4_ln3_t<<<32768, 256, 0, stream>>>(ybuf, ln3_g, ln3_b, u_t);
  k5_sgu<<<dim3(8, 2, 192), 256, 0, stream>>>(u_t, Wsgut, bsgu, ybuf, pbuf);
  k6_stage2<<<dim3(8, 256), 256, 0, stream>>>(pbuf, W2t, b2, xb, x, out, glob);

  (void)in_sizes; (void)n_in; (void)out_size; (void)ws_size;
}
